// PointPillarsEncoder_62096637165640
// MI455X (gfx1250) — matrix operand order repaired and checked by reading, not compiled, not run on hardware
//
#include <hip/hip_runtime.h>

typedef __attribute__((ext_vector_type(16))) __bf16 v16bf;
typedef __attribute__((ext_vector_type(8)))  __bf16 v8bf;
typedef __attribute__((ext_vector_type(8)))  float  v8f;

#define GRID   750
#define XT     47            // ceil(750/16)
#define BATCH  2
#define NVOX   40000
#define PPTS   32
#define HID    64
#define OUTC   128
#define NROWS  (2L * NVOX * PPTS)         // 2,560,000 flat points
#define NPILL  (BATCH * NVOX)             // 80,000
#define NPIX   (2L * GRID * GRID)         // 1,125,000 output pixels
#define NSLOT  1024                        // stats partial slots

// ---------------------------------------------------------------------------
// helpers
// ---------------------------------------------------------------------------
__device__ __forceinline__ void zero_acc(v8f* acc, int n) {
  for (int t = 0; t < n; ++t)
    for (int i = 0; i < 8; ++i) acc[t][i] = 0.f;
}

__device__ __forceinline__ v16bf cat16(v8bf lo, v8bf hi) {
  return __builtin_shufflevector(lo, hi, 0, 1, 2, 3, 4, 5, 6, 7,
                                 8, 9, 10, 11, 12, 13, 14, 15);
}

// Pre-swizzled B-tile fetch: one contiguous 32B load per lane.
// pw layout: [tile][lane][16] bf16, tile = kt*NT + nt
__device__ __forceinline__ v16bf load_btile(const __bf16* __restrict__ pw, int tile) {
  const int lane = threadIdx.x & 31;
  return *(const v16bf*)(pw + ((tile * 32 + lane) << 4));
}

// A-tile (16x32 bf16) from padded aug rows [N,16]; K=16..31 zero.
// D = aug_tile x W1 (K-tile 0 only), 4 N-tiles of 16. pw1: 4 swizzled tiles.
__device__ __forceinline__ void mlp1_tile(const __bf16* __restrict__ aug16,
                                          const __bf16* __restrict__ pw1,
                                          long row0, v8f acc[4]) {
  const int lane = threadIdx.x & 31;
  const int r    = lane & 15;
  const int kbA  = (lane >> 4) * 8;          // A layout: K halves per lane group
  const v8bf lo  = *(const v8bf*)(aug16 + (row0 + r) * 16 + kbA);
  const v8bf zz  = {};
  const v16bf a  = cat16(lo, zz);
#pragma unroll
  for (int nt = 0; nt < 4; ++nt) {
    const v16bf b = load_btile(pw1, nt);
    acc[nt] = __builtin_amdgcn_wmma_f32_16x16x32_bf16(
        false, a, false, b, (short)0, acc[nt], false, false);
  }
}

// y1 -> bn1 affine -> relu -> bf16 LDS tile [16][64] (row-major)
__device__ __forceinline__ void store_h1(const v8f acc[4],
                                         const float* __restrict__ b1,
                                         const float* __restrict__ aff1,
                                         __bf16* __restrict__ h1) {
  const int lane = threadIdx.x & 31;
  const int col0 = lane & 15;
  const int mb   = (lane >> 4) * 8;          // D layout rows
#pragma unroll
  for (int nt = 0; nt < 4; ++nt) {
    const int   c  = nt * 16 + col0;
    const float A  = aff1[c * 2 + 0];
    const float Bo = aff1[c * 2 + 1];
    const float bb = b1[c];
#pragma unroll
    for (int i = 0; i < 8; ++i) {
      float v = (acc[nt][i] + bb) * A + Bo;
      v = v > 0.f ? v : 0.f;
      h1[(mb + i) * HID + c] = (__bf16)v;
    }
  }
}

// h1(LDS,16x64) x W2(64x64) : 2 K-tiles x 4 N-tiles; pw2: 8 swizzled tiles
__device__ __forceinline__ void mlp2_tile(const __bf16* __restrict__ h1,
                                          const __bf16* __restrict__ pw2,
                                          v8f acc[4]) {
  const int lane = threadIdx.x & 31;
  const int m    = lane & 15;
  const int kbA  = (lane >> 4) * 8;
#pragma unroll
  for (int kt = 0; kt < 2; ++kt) {
    const __bf16* ap = h1 + m * HID + kt * 32;
    const v8bf lo = *(const v8bf*)(ap + kbA);
    const v8bf hi = *(const v8bf*)(ap + 16 + kbA);
    const v16bf a = cat16(lo, hi);
#pragma unroll
    for (int nt = 0; nt < 4; ++nt) {
      const v16bf b = load_btile(pw2, kt * 4 + nt);
      acc[nt] = __builtin_amdgcn_wmma_f32_16x16x32_bf16(
          false, a, false, b, (short)0, acc[nt], false, false);
    }
  }
}

// per-channel sum/sumsq of (acc + bias) over the 16-row tile into LDS
__device__ __forceinline__ void stats_accum(const v8f acc[4],
                                            const float* __restrict__ bias,
                                            float* __restrict__ ssum,
                                            float* __restrict__ ssq) {
  const int lane = threadIdx.x & 31;
  const int col0 = lane & 15;
#pragma unroll
  for (int nt = 0; nt < 4; ++nt) {
    const int   c  = nt * 16 + col0;
    const float bb = bias[c];
    float s = 0.f, q = 0.f;
#pragma unroll
    for (int i = 0; i < 8; ++i) { const float v = acc[nt][i] + bb; s += v; q += v * v; }
    const float so = __shfl_xor(s, 16);
    const float qo = __shfl_xor(q, 16);
    if (lane < 16) { atomicAdd(&ssum[c], s + so); atomicAdd(&ssq[c], q + qo); }
  }
}

// ---------------------------------------------------------------------------
// weight prep: swizzle into per-lane WMMA B layout
//   pw[((kt*NT + nt)*32 + lane)*16 + e] = W[kt*32 + (lane>>4)*8 + (e&7) + (e>>3)*16][nt*16 + (lane&15)]
// ---------------------------------------------------------------------------
__global__ void k_prep_w1p(const float* __restrict__ W1, __bf16* __restrict__ pw) {
  const int i = blockIdx.x * 256 + threadIdx.x;
  if (i >= 4 * 32 * 16) return;                 // 1 K-tile, 4 N-tiles
  const int e    = i & 15;
  const int lane = (i >> 4) & 31;
  const int nt   = i >> 9;
  const int K    = ((lane >> 4) * 8) + (e & 7) + ((e >> 3) * 16);
  const int col  = nt * 16 + (lane & 15);
  pw[i] = (__bf16)((K < 10) ? W1[K * HID + col] : 0.f);
}

__global__ void k_prep_w2p(const float* __restrict__ W2, __bf16* __restrict__ pw) {
  const int i = blockIdx.x * 256 + threadIdx.x;
  if (i >= 2 * 4 * 32 * 16) return;             // 2 K-tiles, 4 N-tiles
  const int e    = i & 15;
  const int lane = (i >> 4) & 31;
  const int t    = i >> 9;
  const int nt   = t & 3;
  const int kt   = t >> 2;
  const int K    = kt * 32 + ((lane >> 4) * 8) + (e & 7) + ((e >> 3) * 16);
  const int col  = nt * 16 + (lane & 15);
  pw[i] = (__bf16)W2[K * HID + col];
}

// Wc [Cout,Cin=64,3,3] -> swizzled [18 K-tiles][NT N-tiles][32 lanes][16]
__global__ void k_prep_wcp(const float* __restrict__ Wc, __bf16* __restrict__ pw,
                           int Cout) {
  const int NTt = Cout / 16;
  const int i = blockIdx.x * 256 + threadIdx.x;
  if (i >= 18 * NTt * 32 * 16) return;
  const int e    = i & 15;
  const int lane = (i >> 4) & 31;
  const int t    = i >> 9;
  const int nt   = t % NTt;
  const int kt   = t / NTt;
  const int K    = kt * 32 + ((lane >> 4) * 8) + (e & 7) + ((e >> 3) * 16);  // 0..575
  const int ci   = K & 63;
  const int tap  = K >> 6;                            // ky*3+kx
  const int ky   = tap / 3, kx = tap % 3;
  const int co   = nt * 16 + (lane & 15);
  pw[i] = (__bf16)Wc[((co * HID + ci) * 3 + ky) * 3 + kx];
}

// one wave per pillar: center, offsets, padded bf16 aug rows [N,16]
__global__ void k_pillar_prep(const float* __restrict__ voxels,
                              const int* __restrict__ num_points,
                              __bf16* __restrict__ aug16, int npillars) {
  const int pillar = blockIdx.x;
  if (pillar >= npillars) return;
  const int p = threadIdx.x;                       // 0..31
  const float* vox = voxels + ((long)pillar * PPTS + p) * 4;
  const float v0 = vox[0], v1 = vox[1], v2 = vox[2], v3 = vox[3];
  const int   np  = num_points[pillar];
  const float msk = (p < np) ? 1.f : 0.f;
  float sx = v0 * msk, sy = v1 * msk, sz = v2 * msk, cnt = msk;
#pragma unroll
  for (int off = 16; off > 0; off >>= 1) {
    sx += __shfl_xor(sx, off);  sy += __shfl_xor(sy, off);
    sz += __shfl_xor(sz, off);  cnt += __shfl_xor(cnt, off);
  }
  const float c  = fmaxf(cnt, 1.f);
  const float ox = v0 - sx / c, oy = v1 - sy / c, oz = v2 - sz / c;
  __bf16* row = aug16 + ((long)pillar * PPTS + p) * 16;
  row[0] = (__bf16)v0; row[1] = (__bf16)v1; row[2] = (__bf16)v2; row[3] = (__bf16)v3;
  row[4] = (__bf16)ox; row[5] = (__bf16)oy; row[6] = (__bf16)oz;
  row[7] = (__bf16)ox; row[8] = (__bf16)oy; row[9] = (__bf16)oz;
#pragma unroll
  for (int j = 10; j < 16; ++j) row[j] = (__bf16)0.0f;
}

// ---------------------------------------------------------------------------
// MLP passes
// ---------------------------------------------------------------------------
__global__ void k_gemm1_stats(const __bf16* __restrict__ aug16,
                              const __bf16* __restrict__ pw1,
                              const float* __restrict__ b1,
                              float* __restrict__ partials, long nrows) {
  __shared__ float ssum[HID], ssq[HID];
  const int tid = threadIdx.x;
  if (tid < HID) { ssum[tid] = 0.f; ssq[tid] = 0.f; }
  __syncthreads();
  const long tile = (long)blockIdx.x * 8 + (tid >> 5);
  if (tile * 16 < nrows) {
    v8f acc[4]; zero_acc(acc, 4);
    mlp1_tile(aug16, pw1, tile * 16, acc);
    stats_accum(acc, b1, ssum, ssq);
  }
  __syncthreads();
  if (tid < HID) {
    const int slot = (int)(blockIdx.x & (NSLOT - 1));
    atomicAdd(&partials[(slot * 128 + tid) * 2 + 0], ssum[tid]);
    atomicAdd(&partials[(slot * 128 + tid) * 2 + 1], ssq[tid]);
  }
}

__global__ void k_gemm2_stats(const __bf16* __restrict__ aug16,
                              const __bf16* __restrict__ pw1,
                              const float* __restrict__ b1,
                              const float* __restrict__ aff1,
                              const __bf16* __restrict__ pw2,
                              const float* __restrict__ b2,
                              float* __restrict__ partials, long nrows) {
  __shared__ __attribute__((aligned(16))) __bf16 h1sh[8][16 * HID];
  __shared__ float ssum[HID], ssq[HID];
  const int tid = threadIdx.x;
  if (tid < HID) { ssum[tid] = 0.f; ssq[tid] = 0.f; }
  __syncthreads();
  const int  wave = tid >> 5;
  const long tile = (long)blockIdx.x * 8 + wave;
  if (tile * 16 < nrows) {
    v8f a1[4]; zero_acc(a1, 4);
    mlp1_tile(aug16, pw1, tile * 16, a1);
    store_h1(a1, b1, aff1, &h1sh[wave][0]);
    v8f a2[4]; zero_acc(a2, 4);
    mlp2_tile(&h1sh[wave][0], pw2, a2);
    stats_accum(a2, b2, ssum, ssq);
  }
  __syncthreads();
  if (tid < HID) {
    const int slot = (int)(blockIdx.x & (NSLOT - 1));
    atomicAdd(&partials[(slot * 128 + tid) * 2 + 0], ssum[tid]);
    atomicAdd(&partials[(slot * 128 + tid) * 2 + 1], ssq[tid]);
  }
}

// full recompute + bn2 + relu + mask + max over 32 points; one pillar per wave
__global__ void k_feats(const __bf16* __restrict__ aug16,
                        const __bf16* __restrict__ pw1,
                        const float* __restrict__ b1,
                        const float* __restrict__ aff1,
                        const __bf16* __restrict__ pw2,
                        const float* __restrict__ b2,
                        const float* __restrict__ aff2,
                        const int* __restrict__ num_points,
                        __bf16* __restrict__ pf, int npillars) {
  __shared__ __attribute__((aligned(16))) __bf16 h1sh[8][16 * HID];
  const int tid = threadIdx.x;
  const int wave = tid >> 5, lane = tid & 31;
  const int pillar = blockIdx.x * 8 + wave;
  if (pillar >= npillars) return;                 // wave-uniform exit, no barriers below
  const int np   = num_points[pillar];
  const int col0 = lane & 15;
  const int mb   = (lane >> 4) * 8;
  float cmax[4] = {0.f, 0.f, 0.f, 0.f};          // relu*mask max includes 0
  for (int st = 0; st < 2; ++st) {
    const long row0 = (long)pillar * PPTS + st * 16;
    v8f a1[4]; zero_acc(a1, 4);
    mlp1_tile(aug16, pw1, row0, a1);
    store_h1(a1, b1, aff1, &h1sh[wave][0]);
    v8f a2[4]; zero_acc(a2, 4);
    mlp2_tile(&h1sh[wave][0], pw2, a2);
#pragma unroll
    for (int nt = 0; nt < 4; ++nt) {
      const int   c  = nt * 16 + col0;
      const float A  = aff2[c * 2 + 0];
      const float Bo = aff2[c * 2 + 1];
      const float bb = b2[c];
#pragma unroll
      for (int i = 0; i < 8; ++i) {
        const int p = st * 16 + mb + i;
        float v = (a2[nt][i] + bb) * A + Bo;
        v = v > 0.f ? v : 0.f;
        if (p < np && v > cmax[nt]) cmax[nt] = v;
      }
    }
  }
#pragma unroll
  for (int nt = 0; nt < 4; ++nt) {
    const float o = __shfl_xor(cmax[nt], 16);
    cmax[nt] = cmax[nt] > o ? cmax[nt] : o;
  }
  if (lane < 16) {
#pragma unroll
    for (int nt = 0; nt < 4; ++nt)
      pf[(long)pillar * HID + nt * 16 + col0] = (__bf16)cmax[nt];
  }
}

// ---------------------------------------------------------------------------
// scatter to NHWC bf16 BEV
// ---------------------------------------------------------------------------
__global__ void k_scatter(const __bf16* __restrict__ pf,
                          const int* __restrict__ coords,
                          __bf16* __restrict__ bev, int V) {
  const int bv = blockIdx.x;
  const int b  = bv / V;
  const int x  = coords[(long)bv * 3 + 0];
  const int y  = coords[(long)bv * 3 + 1];
  if (x >= 0 && x < GRID && y >= 0 && y < GRID) {
    const long dst = (((long)b * GRID + y) * GRID + x) * HID + threadIdx.x;
    bev[dst] = pf[(long)bv * HID + threadIdx.x];
  }
}

// ---------------------------------------------------------------------------
// implicit-GEMM 3x3 conv, bf16 WMMA, pre-BN output + channel stats
// ---------------------------------------------------------------------------
template <int NT>
__global__ void k_conv(const __bf16* __restrict__ in,   // [B,750,750,64] bf16 NHWC
                       const __bf16* __restrict__ pwg,  // swizzled [18][NT][32][16]
                       const float* __restrict__ bias,
                       float* __restrict__ partials,
                       __bf16* __restrict__ out_bf,     // NHWC pre-BN (conv1/2)
                       float* __restrict__ out_f32) {   // NCHW pre-BN (conv3 -> d_out)
  constexpr int Cout = NT * 16;
  __shared__ float ssum[Cout], ssq[Cout];
  const int tid = threadIdx.x;
  for (int i = tid; i < Cout; i += 256) { ssum[i] = 0.f; ssq[i] = 0.f; }
  __syncthreads();
  const int  lane   = tid & 31;
  const long tile   = (long)blockIdx.x * 8 + (tid >> 5);
  const long ntiles = (long)BATCH * GRID * XT;
  if (tile < ntiles) {
    const int xt = (int)(tile % XT);
    const int y  = (int)((tile / XT) % GRID);
    const int b  = (int)(tile / ((long)XT * GRID));
    const int x0 = xt * 16;
    const int m  = lane & 15;
    const int x  = x0 + m;
    const int kbA  = (lane >> 4) * 8;
    const int col0 = lane & 15;
    v8f acc[NT]; zero_acc(acc, NT);
#pragma unroll
    for (int ky = 0; ky < 3; ++ky) {
      const int  iy  = y + ky - 1;
      const bool yok = ((unsigned)iy < (unsigned)GRID);
#pragma unroll
      for (int kx = 0; kx < 3; ++kx) {
        const int  ix = x + kx - 1;
        const bool ok = yok && ((unsigned)ix < (unsigned)GRID);
        const __bf16* src = in + (((long)b * GRID + iy) * GRID + ix) * HID;
#pragma unroll
        for (int ct = 0; ct < 2; ++ct) {
          const int ci0 = ct * 32;
          v8bf lo = {}, hi = {};
          if (ok) {
            lo = *(const v8bf*)(src + ci0 + kbA);
            hi = *(const v8bf*)(src + ci0 + 16 + kbA);
          }
          const v16bf a = cat16(lo, hi);
          const int kt = (ky * 3 + kx) * 2 + ct;     // 0..17
#pragma unroll
          for (int nt = 0; nt < NT; ++nt) {
            const v16bf bm = load_btile(pwg, kt * NT + nt);
            acc[nt] = __builtin_amdgcn_wmma_f32_16x16x32_bf16(
                false, a, false, bm, (short)0, acc[nt], false, false);
          }
        }
      }
    }
    const int mb = (lane >> 4) * 8;
#pragma unroll
    for (int nt = 0; nt < NT; ++nt) {
      const int   c  = nt * 16 + col0;
      const float bb = bias[c];
      float s = 0.f, q = 0.f;
#pragma unroll
      for (int i = 0; i < 8; ++i) {
        const int px = x0 + mb + i;
        const float v = acc[nt][i] + bb;
        if (px < GRID) {
          s += v; q += v * v;
          if (out_bf)
            out_bf[(((long)b * GRID + y) * GRID + px) * Cout + c] = (__bf16)v;
          else
            out_f32[(((long)b * Cout + c) * GRID + y) * GRID + px] = v;
        }
      }
      const float so = __shfl_xor(s, 16);
      const float qo = __shfl_xor(q, 16);
      if (lane < 16) { atomicAdd(&ssum[c], s + so); atomicAdd(&ssq[c], q + qo); }
    }
  }
  __syncthreads();
  for (int i = tid; i < Cout; i += 256) {
    const int slot = (int)(blockIdx.x & (NSLOT - 1));
    atomicAdd(&partials[(slot * 128 + i) * 2 + 0], ssum[i]);
    atomicAdd(&partials[(slot * 128 + i) * 2 + 1], ssq[i]);
  }
}

// ---------------------------------------------------------------------------
// stats finalize + elementwise normalizers + zero-fills
// ---------------------------------------------------------------------------
__global__ void k_finalize(const float* __restrict__ partials,
                           const float* __restrict__ g,
                           const float* __restrict__ be,
                           float* __restrict__ affine, float inv_count, int C) {
  const int c = blockIdx.x * blockDim.x + threadIdx.x;
  if (c >= C) return;
  float s = 0.f, q = 0.f;
  for (int i = 0; i < NSLOT; ++i) {
    s += partials[(i * 128 + c) * 2 + 0];
    q += partials[(i * 128 + c) * 2 + 1];
  }
  const float m    = s * inv_count;
  const float var  = q * inv_count - m * m;
  const float rstd = rsqrtf(var + 1e-5f);
  const float A    = rstd * g[c];
  affine[c * 2 + 0] = A;
  affine[c * 2 + 1] = be[c] - m * A;
}

__global__ void k_norm_bf16(__bf16* __restrict__ buf, const float* __restrict__ aff,
                            long n, int C) {
  const long i = (long)blockIdx.x * 256 + threadIdx.x;
  if (i >= n) return;
  const int c = (int)(i % C);
  const float v = (float)buf[i] * aff[c * 2 + 0] + aff[c * 2 + 1];
  buf[i] = (__bf16)(v > 0.f ? v : 0.f);
}

__global__ void k_norm_f32_nchw(float* __restrict__ buf, const float* __restrict__ aff,
                                long n, long HW, int C) {
  const long i = (long)blockIdx.x * 256 + threadIdx.x;
  if (i >= n) return;
  const int c = (int)((i / HW) % C);
  const float v = buf[i] * aff[c * 2 + 0] + aff[c * 2 + 1];
  buf[i] = v > 0.f ? v : 0.f;
}

__global__ void k_zero32(float* __restrict__ p, long n) {
  const long i = (long)blockIdx.x * 256 + threadIdx.x;
  if (i < n) p[i] = 0.f;
}
__global__ void k_zero16(__bf16* __restrict__ p, long n) {
  const long i = (long)blockIdx.x * 256 + threadIdx.x;
  if (i < n) p[i] = (__bf16)0.0f;
}

// ---------------------------------------------------------------------------
// host launcher
// ---------------------------------------------------------------------------
extern "C" void kernel_launch(void* const* d_in, const int* in_sizes, int n_in,
                              void* d_out, int out_size, void* d_ws, size_t ws_size,
                              hipStream_t stream) {
  (void)in_sizes; (void)n_in; (void)out_size; (void)ws_size;
  const float* voxels = (const float*)d_in[0];
  const int*   num_pts = (const int*)d_in[1];
  const int*   coords  = (const int*)d_in[2];
  const float* W1  = (const float*)d_in[3];
  const float* b1  = (const float*)d_in[4];
  const float* g1  = (const float*)d_in[5];
  const float* be1 = (const float*)d_in[6];
  const float* W2  = (const float*)d_in[7];
  const float* b2  = (const float*)d_in[8];
  const float* g2  = (const float*)d_in[9];
  const float* be2 = (const float*)d_in[10];
  const float* Wc1 = (const float*)d_in[11];
  const float* bc1 = (const float*)d_in[12];
  const float* gc1 = (const float*)d_in[13];
  const float* bec1 = (const float*)d_in[14];
  const float* Wc2 = (const float*)d_in[15];
  const float* bc2 = (const float*)d_in[16];
  const float* gc2 = (const float*)d_in[17];
  const float* bec2 = (const float*)d_in[18];
  const float* Wc3 = (const float*)d_in[19];
  const float* bc3 = (const float*)d_in[20];
  const float* gc3 = (const float*)d_in[21];
  const float* bec3 = (const float*)d_in[22];
  float* out = (float*)d_out;

  char* p = (char*)d_ws;
  auto alloc = [&](size_t bytes) -> void* {
    void* r = (void*)p;
    p += (bytes + 255) & ~(size_t)255;
    return r;
  };
  __bf16* aug16 = (__bf16*)alloc((size_t)NROWS * 16 * 2);
  __bf16* pw1   = (__bf16*)alloc((size_t)4 * 32 * 16 * 2);           // 1 KT x 4 NT
  __bf16* pw2   = (__bf16*)alloc((size_t)8 * 32 * 16 * 2);           // 2 KT x 4 NT
  __bf16* pwg1  = (__bf16*)alloc((size_t)18 * 4 * 32 * 16 * 2);      // 18 KT x 4 NT
  __bf16* pwg2  = (__bf16*)alloc((size_t)18 * 4 * 32 * 16 * 2);
  __bf16* pwg3  = (__bf16*)alloc((size_t)18 * 8 * 32 * 16 * 2);      // 18 KT x 8 NT
  float*  part  = (float*)alloc((size_t)NSLOT * 128 * 2 * 4);
  float*  aff1  = (float*)alloc(HID * 2 * 4);
  float*  aff2  = (float*)alloc(HID * 2 * 4);
  float*  affc1 = (float*)alloc(HID * 2 * 4);
  float*  affc2 = (float*)alloc(HID * 2 * 4);
  float*  affc3 = (float*)alloc(OUTC * 2 * 4);
  __bf16* pf    = (__bf16*)alloc((size_t)NPILL * HID * 2);
  __bf16* bev   = (__bf16*)alloc((size_t)NPIX * HID * 2);
  __bf16* buf1  = (__bf16*)alloc((size_t)NPIX * HID * 2);
  __bf16* buf2  = (__bf16*)alloc((size_t)NPIX * HID * 2);

  const long partN = (long)NSLOT * 128 * 2;
  const long bevN  = NPIX * HID;            // 72,000,000
  const long outN  = NPIX * OUTC;           // 144,000,000

  // weights (swizzled into WMMA B-layout)
  k_prep_w1p<<<8, 256, 0, stream>>>(W1, pw1);
  k_prep_w2p<<<16, 256, 0, stream>>>(W2, pw2);
  k_prep_wcp<<<(18 * 4 * 512 + 255) / 256, 256, 0, stream>>>(Wc1, pwg1, HID);
  k_prep_wcp<<<(18 * 4 * 512 + 255) / 256, 256, 0, stream>>>(Wc2, pwg2, HID);
  k_prep_wcp<<<(18 * 8 * 512 + 255) / 256, 256, 0, stream>>>(Wc3, pwg3, OUTC);

  // pillar features
  k_pillar_prep<<<NPILL, 32, 0, stream>>>(voxels, num_pts, aug16, NPILL);

  const int mlpBlocks = (int)(NROWS / 16 / 8);       // 20000
  // BN1 stats
  k_zero32<<<(int)((partN + 255) / 256), 256, 0, stream>>>(part, partN);
  k_gemm1_stats<<<mlpBlocks, 256, 0, stream>>>(aug16, pw1, b1, part, NROWS);
  k_finalize<<<1, 64, 0, stream>>>(part, g1, be1, aff1, 1.f / (float)NROWS, HID);
  // BN2 stats
  k_zero32<<<(int)((partN + 255) / 256), 256, 0, stream>>>(part, partN);
  k_gemm2_stats<<<mlpBlocks, 256, 0, stream>>>(aug16, pw1, b1, aff1, pw2, b2, part, NROWS);
  k_finalize<<<1, 64, 0, stream>>>(part, g2, be2, aff2, 1.f / (float)NROWS, HID);
  // pillar max-pool + scatter
  k_feats<<<NPILL / 8, 256, 0, stream>>>(aug16, pw1, b1, aff1, pw2, b2, aff2,
                                         num_pts, pf, NPILL);
  k_zero16<<<(int)((bevN + 255) / 256), 256, 0, stream>>>(bev, bevN);
  k_scatter<<<NPILL, 64, 0, stream>>>(pf, coords, bev, NVOX);

  // convs
  const long ntiles = (long)BATCH * GRID * XT;       // 70500
  const int convBlocks = (int)((ntiles + 7) / 8);    // 8813
  const float invPix = 1.f / (float)NPIX;

  k_zero32<<<(int)((partN + 255) / 256), 256, 0, stream>>>(part, partN);
  k_conv<4><<<convBlocks, 256, 0, stream>>>(bev, pwg1, bc1, part, buf1, nullptr);
  k_finalize<<<1, 64, 0, stream>>>(part, gc1, bec1, affc1, invPix, HID);
  k_norm_bf16<<<(int)((bevN + 255) / 256), 256, 0, stream>>>(buf1, affc1, bevN, HID);

  k_zero32<<<(int)((partN + 255) / 256), 256, 0, stream>>>(part, partN);
  k_conv<4><<<convBlocks, 256, 0, stream>>>(buf1, pwg2, bc2, part, buf2, nullptr);
  k_finalize<<<1, 64, 0, stream>>>(part, gc2, bec2, affc2, invPix, HID);
  k_norm_bf16<<<(int)((bevN + 255) / 256), 256, 0, stream>>>(buf2, affc2, bevN, HID);

  k_zero32<<<(int)((partN + 255) / 256), 256, 0, stream>>>(part, partN);
  k_conv<8><<<convBlocks, 256, 0, stream>>>(buf2, pwg3, bc3, part, nullptr, out);
  k_finalize<<<1, 128, 0, stream>>>(part, gc3, bec3, affc3, invPix, OUTC);
  k_norm_f32_nchw<<<(int)((outN + 255) / 256), 256, 0, stream>>>(out, affc3, outN,
                                                                 (long)GRID * GRID, OUTC);
}
